// Attention_35820027248943
// MI455X (gfx1250) — matrix operand order repaired and checked by reading, not compiled, not run on hardware
//
#include <hip/hip_runtime.h>
#include <hip/hip_bf16.h>

// ---- problem constants (from reference) ----
#define DIMC    768
#define NHEADS  12
#define HDIM    64
#define SEQ     2048
#define BATCH   4
#define MTOT    (BATCH*SEQ)          // 8192 rows of x
#define QKVO    (3*DIMC)             // 2304
#define ATTN_SCALE 0.125f            // 64^-0.5

typedef _Float16 v16h __attribute__((ext_vector_type(16)));
typedef _Float16 v8h  __attribute__((ext_vector_type(8)));
typedef float    v8f  __attribute__((ext_vector_type(8)));
typedef unsigned int v4u __attribute__((ext_vector_type(4)));
typedef int          v4i __attribute__((ext_vector_type(4)));
typedef int          v8i __attribute__((ext_vector_type(8)));

union Frag16 { v16h v; v8h h[2]; };
union U4H8   { uint4 u; _Float16 h[8]; };

#if defined(__has_builtin)
#if __has_builtin(__builtin_amdgcn_tensor_load_to_lds) && \
    __has_builtin(__builtin_amdgcn_s_wait_tensorcnt)
#define USE_TDM 1
#endif
#endif

#ifdef USE_TDM
// Tensor Data Mover: DMA a 2-D tile of f16 (tile_w x tile_h elements, global
// row stride `row_stride` elements) from global memory into contiguous LDS.
// D# bitfields per CDNA5 ISA ch.8 (group0: count/lds_addr/global_addr/type,
// group1: data_size=2B, tensor dims == tile dims (no OOB), dim0 stride).
// This toolchain exposes the 6-arg builtin:
//   (uint32x4 g0, int32x8 g1, int32x4 g2, int32x4 g3, int32x8 g4, i32 cpol)
__device__ __forceinline__ void tdm_load_2d_f16(unsigned lds_addr,
                                                const void* gptr,
                                                unsigned tile_w, unsigned tile_h,
                                                unsigned row_stride) {
  unsigned long long ga = (unsigned long long)(uintptr_t)gptr;
  v4u g0;
  g0[0] = 1u;                                        // count=1, user mode
  g0[1] = lds_addr;                                  // LDS byte address
  g0[2] = (unsigned)(ga & 0xffffffffu);              // global_addr[31:0]
  g0[3] = (unsigned)((ga >> 32) & 0x1ffffffu)        // global_addr[56:32]
        | 0x80000000u;                               // type=2 ("image")
  v8i g1;
  g1[0] = (int)(1u << 16);                           // data_size=1 -> 2 bytes
  g1[1] = (int)((tile_w & 0xffffu) << 16);           // tensor_dim0[15:0]
  g1[2] = (int)(((tile_w >> 16) & 0xffffu)           // tensor_dim0[31:16]
        | ((tile_h & 0xffffu) << 16));               // tensor_dim1[15:0]
  g1[3] = (int)(((tile_h >> 16) & 0xffffu)           // tensor_dim1[31:16]
        | ((tile_w & 0xffffu) << 16));               // tile_dim0
  g1[4] = (int)(tile_h & 0xffffu);                   // tile_dim1 (tile_dim2=0)
  g1[5] = (int)row_stride;                           // tensor_dim0_stride[31:0]
  g1[6] = 0;                                         // stride hi / dim1_stride lo
  g1[7] = 0;
  v4i z4 = {0, 0, 0, 0};                             // groups 2/3: 2-D tensor
  v8i z8 = {0, 0, 0, 0, 0, 0, 0, 0};
  __builtin_amdgcn_tensor_load_to_lds(g0, g1, z4, z4, z8, 0);
}
#endif

__device__ __forceinline__ unsigned lds_off(const void* p) {
  return (unsigned)(uintptr_t)p;   // low 32 bits of flat addr == LDS byte offset
}

// -------------------- fp32 -> fp16 converter --------------------
__global__ void f32_to_f16_kernel(const float* __restrict__ src,
                                  _Float16* __restrict__ dst, int n) {
  int i = blockIdx.x * blockDim.x + threadIdx.x;
  int stride = gridDim.x * blockDim.x;
  for (; i < n; i += stride) dst[i] = (_Float16)src[i];
}

// -------------------- tiled WMMA GEMM --------------------
// C[m, n] = sum_k A[m,k] * Bw[n,k] + bias[n]
// MODE 0: epilogue scatters into q/k/v (f16, [B,H,N,64]); q pre-scaled by ATTN_SCALE
// MODE 1: epilogue writes fp32 out[m*Ncols + n]
template <int MODE>
__global__ void __launch_bounds__(256)
gemm_wmma(const _Float16* __restrict__ A, const _Float16* __restrict__ Bw,
          const float* __restrict__ bias,
          _Float16* __restrict__ qh, _Float16* __restrict__ kh,
          _Float16* __restrict__ vh,
          float* __restrict__ outf, int Ncols, int Kdim) {
  __shared__ _Float16 Asl[128][32];   // 8 KB, row stride 64B
  __shared__ _Float16 Bsl[128][32];   // 8 KB

  const int tid  = threadIdx.x;
  const int lane = tid & 31;
  const int w    = tid >> 5;      // wave 0..7
  const int wm   = w & 3;         // 4 M-subtiles of 32
  const int wn   = w >> 2;        // 2 N-subtiles of 64
  const int m0   = blockIdx.x * 128;
  const int n0   = blockIdx.y * 128;

  v8f acc[2][4];
#pragma unroll
  for (int i = 0; i < 2; i++)
#pragma unroll
    for (int j = 0; j < 4; j++) acc[i][j] = {};

#ifdef USE_TDM
  const unsigned aOff = lds_off(&Asl[0][0]);
  const unsigned bOff = lds_off(&Bsl[0][0]);
#else
  const int ldrow = tid >> 1;           // 0..127
  const int ldseg = (tid & 1) * 16;     // 0 or 16 halves (32B each)
#endif

  for (int k0 = 0; k0 < Kdim; k0 += 32) {
    __syncthreads();
#ifdef USE_TDM
    if (w == 0) {   // one wave drives the Tensor Data Mover for both tiles
      tdm_load_2d_f16(aOff, A  + (size_t)m0 * Kdim + k0, 32, 128, (unsigned)Kdim);
      tdm_load_2d_f16(bOff, Bw + (size_t)n0 * Kdim + k0, 32, 128, (unsigned)Kdim);
      __builtin_amdgcn_s_wait_tensorcnt(0);
    }
#else
    {
      const uint4* asrc = (const uint4*)(A + (size_t)(m0 + ldrow) * Kdim + k0 + ldseg);
      uint4* adst = (uint4*)&Asl[ldrow][ldseg];
      adst[0] = asrc[0]; adst[1] = asrc[1];
      const uint4* bsrc = (const uint4*)(Bw + (size_t)(n0 + ldrow) * Kdim + k0 + ldseg);
      uint4* bdst = (uint4*)&Bsl[ldrow][ldseg];
      bdst[0] = bsrc[0]; bdst[1] = bsrc[1];
    }
#endif
    __syncthreads();

    const int ahi = (lane < 16) ? 0 : 8;     // A-layout K chunk select
    const int bhi = (lane < 16) ? 0 : 8;     // B-layout K chunk select

    Frag16 afr[2], bfr[4];
#pragma unroll
    for (int mt = 0; mt < 2; mt++) {
      int r = wm * 32 + mt * 16 + (lane & 15);
      afr[mt].h[0] = *(const v8h*)&Asl[r][ahi];        // K = {0..7} or {8..15}
      afr[mt].h[1] = *(const v8h*)&Asl[r][16 + ahi];   // K = {16..23} or {24..31}
    }
#pragma unroll
    for (int nt = 0; nt < 4; nt++) {
      int c = wn * 64 + nt * 16 + (lane & 15);
      bfr[nt].h[0] = *(const v8h*)&Bsl[c][bhi];
      bfr[nt].h[1] = *(const v8h*)&Bsl[c][bhi + 16];
    }
#pragma unroll
    for (int mt = 0; mt < 2; mt++)
#pragma unroll
      for (int nt = 0; nt < 4; nt++)
        acc[mt][nt] = __builtin_amdgcn_wmma_f32_16x16x32_f16(
            false, afr[mt].v, false, bfr[nt].v, (short)0, acc[mt][nt], false, false);
  }

  // ---- epilogue ----
  const int colb = n0 + wn * 64 + (lane & 15);
  const int rowb = m0 + wm * 32 + ((lane < 16) ? 0 : 8);
#pragma unroll
  for (int mt = 0; mt < 2; mt++) {
#pragma unroll
    for (int nt = 0; nt < 4; nt++) {
      const int col = colb + nt * 16;
      const float bval = bias[col];
#pragma unroll
      for (int r = 0; r < 8; r++) {
        const int m = rowb + mt * 16 + r;
        float v = acc[mt][nt][r] + bval;
        if (MODE == 0) {
          int which = col / DIMC;
          int rem   = col - which * DIMC;
          int hh    = rem >> 6, d = rem & 63;
          int b     = m >> 11, s = m & 2047;
          size_t idx = ((size_t)((b * NHEADS + hh) * SEQ + s)) * HDIM + d;
          if (which == 0)      qh[idx] = (_Float16)(v * ATTN_SCALE);
          else if (which == 1) kh[idx] = (_Float16)v;
          else                 vh[idx] = (_Float16)v;
        } else {
          outf[(size_t)m * Ncols + col] = v;
        }
      }
    }
  }
}

// -------------------- flash attention --------------------
// grid.x = B*H (48), grid.y = SEQ/128; 8 waves/block, 16 queries/wave
__global__ void __launch_bounds__(256)
flash_attn(const _Float16* __restrict__ qh, const _Float16* __restrict__ kh,
           const _Float16* __restrict__ vh, _Float16* __restrict__ oh) {
  __shared__ _Float16 Ksl[32][64];       // 4 KB, [key][d]   (TDM destination)
  __shared__ _Float16 Vts[64][32];       // 4 KB, [d][key]   (transposed)
  __shared__ _Float16 Psl[8][16][32];    // 8 KB, per-wave P staging

  const int tid  = threadIdx.x;
  const int lane = tid & 31;
  const int w    = tid >> 5;
  const int bh   = blockIdx.x;                    // b*NHEADS + h
  const int q0   = blockIdx.y * 128 + w * 16;
  const size_t base = (size_t)bh * SEQ * HDIM;
#ifdef USE_TDM
  const unsigned kOff = lds_off(&Ksl[0][0]);
#endif

  // Q fragments (A-layout), scale already folded in at QKV epilogue
  Frag16 qa[2];
  {
    const int qr = q0 + (lane & 15);
    const int ahi = (lane < 16) ? 0 : 8;
#pragma unroll
    for (int j = 0; j < 2; j++) {
      const _Float16* p = qh + base + (size_t)qr * HDIM + j * 32;
      qa[j].h[0] = *(const v8h*)(p + ahi);
      qa[j].h[1] = *(const v8h*)(p + 16 + ahi);
    }
  }

  float m_i[8], l_i[8];
  v8f o[4];
#pragma unroll
  for (int r = 0; r < 8; r++) { m_i[r] = -__builtin_inff(); l_i[r] = 0.f; }
#pragma unroll
  for (int t = 0; t < 4; t++) o[t] = {};

  for (int key0 = 0; key0 < SEQ; key0 += 32) {
    __syncthreads();
    {
      const int krow = tid >> 3;          // 0..31
      const int d0   = (tid & 7) * 8;     // 0..56
      // V: manual transposed store (TDM cannot transpose)
      U4H8 vv;
      vv.u = *(const uint4*)(vh + base + (size_t)(key0 + krow) * HDIM + d0);
#pragma unroll
      for (int i = 0; i < 8; i++) Vts[d0 + i][krow] = vv.h[i];
#ifdef USE_TDM
      if (w == 0) {   // K tile (32 keys x 64 dims, contiguous rows) via TDM
        tdm_load_2d_f16(kOff, kh + base + (size_t)key0 * HDIM, 64, 32, HDIM);
        __builtin_amdgcn_s_wait_tensorcnt(0);
      }
#else
      *(uint4*)&Ksl[krow][d0] =
          *(const uint4*)(kh + base + (size_t)(key0 + krow) * HDIM + d0);
#endif
    }
    __syncthreads();

    const int bhi = (lane < 16) ? 0 : 8;

    // S = Q * K^T  (two 16x16 key tiles, K-dim 64 = 2 WMMAs each)
    v8f s[2];
#pragma unroll
    for (int kt = 0; kt < 2; kt++) {
      const int kr = kt * 16 + (lane & 15);
      v8f c = {};
#pragma unroll
      for (int j = 0; j < 2; j++) {
        Frag16 kb;
        kb.h[0] = *(const v8h*)&Ksl[kr][j * 32 + bhi];
        kb.h[1] = *(const v8h*)&Ksl[kr][j * 32 + bhi + 16];
        c = __builtin_amdgcn_wmma_f32_16x16x32_f16(false, qa[j].v, false, kb.v,
                                                   (short)0, c, false, false);
      }
      s[kt] = c;
    }

    // online softmax over 32 keys; rows live as 8 regs/lane, cols across 16 lanes
    float mx[8], p0[8], p1[8], sum[8];
#pragma unroll
    for (int r = 0; r < 8; r++) mx[r] = fmaxf(s[0][r], s[1][r]);
#pragma unroll
    for (int off = 1; off < 16; off <<= 1)
#pragma unroll
      for (int r = 0; r < 8; r++) mx[r] = fmaxf(mx[r], __shfl_xor(mx[r], off, 32));

#pragma unroll
    for (int r = 0; r < 8; r++) {
      const float mn    = fmaxf(m_i[r], mx[r]);
      const float alpha = __expf(m_i[r] - mn);
      p0[r]  = __expf(s[0][r] - mn);
      p1[r]  = __expf(s[1][r] - mn);
      sum[r] = p0[r] + p1[r];
      l_i[r] *= alpha;
      m_i[r] = mn;
#pragma unroll
      for (int t = 0; t < 4; t++) o[t][r] *= alpha;
    }
#pragma unroll
    for (int off = 1; off < 16; off <<= 1)
#pragma unroll
      for (int r = 0; r < 8; r++) sum[r] += __shfl_xor(sum[r], off, 32);
#pragma unroll
    for (int r = 0; r < 8; r++) l_i[r] += sum[r];

    // P (f32, D-layout) -> LDS -> A-layout f16 fragment
    {
      const int colj  = lane & 15;
      const int rbase = (lane < 16) ? 0 : 8;
#pragma unroll
      for (int r = 0; r < 8; r++) {
        Psl[w][rbase + r][colj]      = (_Float16)p0[r];
        Psl[w][rbase + r][16 + colj] = (_Float16)p1[r];
      }
    }
    Frag16 pa;
    {
      const int pr  = lane & 15;
      const int ahi = (lane < 16) ? 0 : 8;
      pa.h[0] = *(const v8h*)&Psl[w][pr][ahi];
      pa.h[1] = *(const v8h*)&Psl[w][pr][16 + ahi];
    }

    // O += P * V   (4 d-tiles of 16)
#pragma unroll
    for (int t = 0; t < 4; t++) {
      Frag16 vb;
      const int d = t * 16 + (lane & 15);
      vb.h[0] = *(const v8h*)&Vts[d][bhi];
      vb.h[1] = *(const v8h*)&Vts[d][bhi + 16];
      o[t] = __builtin_amdgcn_wmma_f32_16x16x32_f16(false, pa.v, false, vb.v,
                                                    (short)0, o[t], false, false);
    }
  }

  // normalize + store to [B, N, DIMC] f16 for the projection GEMM
  {
    const int b     = bh / NHEADS;
    const int hh    = bh - b * NHEADS;
    const int rbase = (lane < 16) ? 0 : 8;
    const int dcol  = lane & 15;
#pragma unroll
    for (int r = 0; r < 8; r++) {
      const float inv = 1.f / l_i[r];
      const int qrow  = q0 + rbase + r;
      const size_t rowoff = ((size_t)(b * SEQ + qrow)) * DIMC + hh * HDIM;
#pragma unroll
      for (int t = 0; t < 4; t++)
        oh[rowoff + t * 16 + dcol] = (_Float16)(o[t][r] * inv);
    }
  }
}

// -------------------- host launcher --------------------
extern "C" void kernel_launch(void* const* d_in, const int* in_sizes, int n_in,
                              void* d_out, int out_size, void* d_ws, size_t ws_size,
                              hipStream_t stream) {
  (void)in_sizes; (void)n_in; (void)out_size; (void)ws_size;
  const float* x      = (const float*)d_in[0];
  const float* w_qkv  = (const float*)d_in[1];
  const float* b_qkv  = (const float*)d_in[2];
  const float* w_proj = (const float*)d_in[3];
  const float* b_proj = (const float*)d_in[4];
  float* out = (float*)d_out;

  char* ws = (char*)d_ws;
  size_t off = 0;
  auto carve = [&](size_t bytes) {
    char* p = ws + off;
    off += (bytes + 255) & ~(size_t)255;
    return p;
  };
  const size_t nX    = (size_t)MTOT * DIMC;      // 6.29M halves
  const size_t nWq   = (size_t)QKVO * DIMC;
  const size_t nWp   = (size_t)DIMC * DIMC;
  const size_t nHead = (size_t)BATCH * NHEADS * SEQ * HDIM;   // == nX

  _Float16* xh  = (_Float16*)carve(nX * 2);
  _Float16* wqh = (_Float16*)carve(nWq * 2);
  _Float16* wph = (_Float16*)carve(nWp * 2);
  _Float16* qh  = (_Float16*)carve(nHead * 2);
  _Float16* kh  = (_Float16*)carve(nHead * 2);
  _Float16* vh  = (_Float16*)carve(nHead * 2);
  _Float16* oh  = (_Float16*)carve(nX * 2);

  f32_to_f16_kernel<<<2048, 256, 0, stream>>>(x, xh, (int)nX);
  f32_to_f16_kernel<<<2048, 256, 0, stream>>>(w_qkv, wqh, (int)nWq);
  f32_to_f16_kernel<<<1024, 256, 0, stream>>>(w_proj, wph, (int)nWp);

  // QKV GEMM: 8192 x 2304 x 768
  gemm_wmma<0><<<dim3(MTOT / 128, QKVO / 128), 256, 0, stream>>>(
      xh, wqh, b_qkv, qh, kh, vh, nullptr, QKVO, DIMC);

  // flash attention: 48 heads x 16 query blocks
  flash_attn<<<dim3(BATCH * NHEADS, SEQ / 128), 256, 0, stream>>>(qh, kh, vh, oh);

  // projection GEMM: 8192 x 768 x 768, fp32 out + bias
  gemm_wmma<1><<<dim3(MTOT / 128, DIMC / 128), 256, 0, stream>>>(
      oh, wph, b_proj, nullptr, nullptr, nullptr, out, DIMC, DIMC);
}
